// GptOssAttention_57354993271420
// MI455X (gfx1250) — matrix operand order repaired and checked by reading, not compiled, not run on hardware
//
#include <hip/hip_runtime.h>
#include <cstdint>
#include <cstddef>

// ---------------------------------------------------------------------------
// GPT-OSS attention layer for MI455X (gfx1250, wave32, WMMA bf16 + TDM)
//   1) convert hidden -> bf16 ; transpose+convert w_qkv, w_o -> bf16 (N x K)
//   2) QKV GEMM: bf16 WMMA 16x16x32, TDM double-buffered LDS tiles
//   3) RoPE + split into q/k/v bf16 buffers
//   4) sink-window attention (TDM K-tile, WMMA scores/PV, register softmax)
//   5) output GEMM + bias -> fp32 d_out
// ---------------------------------------------------------------------------

typedef __bf16 bf16_t;
typedef __attribute__((ext_vector_type(16))) __bf16 v16bf;
typedef __attribute__((ext_vector_type(8)))  float  v8f;
typedef unsigned int v4u __attribute__((ext_vector_type(4)));
typedef int          v4i __attribute__((ext_vector_type(4)));
typedef int          v8i __attribute__((ext_vector_type(8)));

#define SEQ   4096
#define HID   2880
#define NQKV  5120        // (64 + 2*8) * 64
#define NQH   64
#define NKVH  8
#define GQA   8           // NQH / NKVH
#define HDIM  64
#define QDIM  4096        // NQH * HDIM
#define WWIN  128
#define ATT_SCALE 0.125f  // 64^-0.5

// LDS strides (elements); multiples of 8 elems (16B) keep ds_load_b128 aligned.
#define KT_STRIDE 80      // K tile rows: 64 elems + 16-elem TDM pad
#define VT_STRIDE 272     // V^T / P rows: 256 keys + 16-elem (32B) pad

union Frag32B { struct { uint4 lo, hi; } u; v16bf v; };

// A-matrix fragment (16x32 bf16) from a row-major [16 x >=32] buffer.
__device__ __forceinline__ v16bf load_a_frag(const bf16_t* base, int stride, int lane) {
  const int r = lane & 15, half = lane >> 4;
  const bf16_t* p = base + (size_t)r * stride + half * 8;
  Frag32B f;
  f.u.lo = *(const uint4*)(p);
  f.u.hi = *(const uint4*)(p + 16);
  return f.v;
}

// B-matrix fragment (32x16 bf16) from B^T storage (rows = N dimension).
__device__ __forceinline__ v16bf load_b_frag(const bf16_t* base, int stride, int lane) {
  const int n = lane & 15, half = lane >> 4;
  const bf16_t* p = base + (size_t)n * stride + half * 8;
  Frag32B f;
  f.u.lo = *(const uint4*)(p);
  f.u.hi = *(const uint4*)(p + 16);
  return f.v;
}

__device__ __forceinline__ v8f wmma_bf16(v16bf a, v16bf b, v8f c) {
  return __builtin_amdgcn_wmma_f32_16x16x32_bf16(false, a, false, b, (short)0, c, false, false);
}

// ---------------------------------------------------------------------------
// TDM: 2D bf16 tile load, global -> LDS, with LDS row padding.
// Builds the D# per CDNA5 ISA:
//   group0: {count=1 | lds_addr | global_addr | type=2}
//   group1: {flags(data_size=2B, pad) | tensor_dim0/1 | tile_dim0/1 | dim0 stride}
// pad_interval_code: pad after 2^(c+1) DWORDs; pad_amount_code: (c+1) DWORDs.
// OOB rows (>= tensor_d1) read as zero (used for the N-edge of the out GEMM).
// ---------------------------------------------------------------------------
__device__ __forceinline__ void tdm_load_2d_bf16(
    unsigned int lds_byte_addr, const void* gptr,
    unsigned int tile_d0, unsigned int tile_d1,
    unsigned long long row_stride_elems,
    unsigned int tensor_d0, unsigned int tensor_d1,
    unsigned int pad_interval_code, unsigned int pad_amount_code) {
  unsigned long long ga = (unsigned long long)(size_t)gptr;
  v4u g0;
  g0.x = 1u;                                                   // count=1, user mode
  g0.y = lds_byte_addr;                                        // D#.lds_addr (bytes)
  g0.z = (unsigned int)ga;                                     // global_addr[31:0]
  g0.w = (unsigned int)((ga >> 32) & 0x01ffffffu) | (2u << 30);// [56:32] | type=2
  const unsigned int flags = (1u << 16)                        // data_size = 2 bytes
                           | (1u << 20)                        // pad_enable
                           | (pad_interval_code << 22)
                           | (pad_amount_code << 25);
  v8i g1;
  g1[0] = (int)flags;
  g1[1] = (int)((tensor_d0 & 0xffffu) << 16);                  // atomic_addr=0 | tdim0.lo
  g1[2] = (int)(((tensor_d0 >> 16) & 0xffffu) | ((tensor_d1 & 0xffffu) << 16));
  g1[3] = (int)(((tensor_d1 >> 16) & 0xffffu) | ((tile_d0 & 0xffffu) << 16));
  g1[4] = (int)(tile_d1 & 0xffffu);                            // tile_dim1 | tile_dim2=0
  g1[5] = (int)(unsigned int)(row_stride_elems & 0xffffffffu); // dim0_stride lo32
  g1[6] = (int)(unsigned int)((row_stride_elems >> 32) & 0xffffu);
  g1[7] = 0;
  v4i z4 = (v4i){0, 0, 0, 0};
#if defined(__clang_major__) && (__clang_major__ >= 23)
  v8i z8 = (v8i){0, 0, 0, 0, 0, 0, 0, 0};
  __builtin_amdgcn_tensor_load_to_lds(g0, g1, z4, z4, z8, 0);
#else
  __builtin_amdgcn_tensor_load_to_lds(g0, g1, z4, z4, 0);
#endif
}

// ---------------------------------------------------------------------------
// fp32 -> bf16 elementwise convert
// ---------------------------------------------------------------------------
__global__ void convert_f32_bf16_kernel(const float* __restrict__ src,
                                        bf16_t* __restrict__ dst, size_t n) {
  size_t i = (size_t)blockIdx.x * blockDim.x + threadIdx.x;
  size_t stride = (size_t)gridDim.x * blockDim.x;
  for (; i < n; i += stride) dst[i] = (bf16_t)src[i];
}

// ---------------------------------------------------------------------------
// Transposing convert: src (R x C, fp32 row-major) -> dst (C x R, bf16)
// ---------------------------------------------------------------------------
__global__ __launch_bounds__(256) void transpose_bf16_kernel(
    const float* __restrict__ src, bf16_t* __restrict__ dst, int R, int C) {
  __shared__ float tile[32][33];
  const int c0 = blockIdx.x * 32, r0 = blockIdx.y * 32;
  const int tx = threadIdx.x & 31, ty = threadIdx.x >> 5;
#pragma unroll
  for (int i = ty; i < 32; i += 8) {
    int r = r0 + i, c = c0 + tx;
    tile[i][tx] = (r < R && c < C) ? src[(size_t)r * C + c] : 0.f;
  }
  __syncthreads();
#pragma unroll
  for (int i = ty; i < 32; i += 8) {
    int c = c0 + i, r = r0 + tx;
    if (c < C && r < R) dst[(size_t)c * R + r] = (bf16_t)tile[tx][i];
  }
}

// ---------------------------------------------------------------------------
// bf16 WMMA GEMM:  C(MxN, f32) = A(MxK bf16) * BT^T + bias(N); BT is (N x K).
// 256 thr = 8 waves; tile 128x128; wave tile 64x32; BK = 32.
// TDM double buffering: wave 0 issues next A/B tiles while current computes.
// NOTE: no local arrays of LDS-derived pointers (lld cannot relocate
// addrspacecast constants in static initializers) -- compute from `cur`.
// ---------------------------------------------------------------------------
#define GBM 128
#define GBN 128
#define GBK 32
#define GSK 40  // 32 elems + 4-DWORD TDM pad -> 80B rows (b128 aligned)
#define GEMM_TILE_ELEMS (GBM * GSK)
#define GEMM_TILE_BYTES (GEMM_TILE_ELEMS * 2)
#define GEMM_SMEM (4 * GEMM_TILE_BYTES)

__global__ __launch_bounds__(256) void gemm_bf16_kernel(
    const bf16_t* __restrict__ A, const bf16_t* __restrict__ BT,
    const float* __restrict__ bias, float* __restrict__ C,
    int M, int N, int K) {
  extern __shared__ bf16_t gsm[];
  const unsigned int lds0 = __builtin_amdgcn_groupstaticsize();

  const int tid  = threadIdx.x;
  const int lane = tid & 31;
  const int w    = tid >> 5;
  const int wr   = w >> 2;   // 0..1  (wave rows of 64)
  const int wc   = w & 3;    // 0..3  (wave cols of 32)
  const int brow = blockIdx.y * GBM;
  const int bcol = blockIdx.x * GBN;
  const unsigned int bn_valid =
      (unsigned int)((N - bcol) < GBN ? (N - bcol) : GBN);  // OOB rows -> 0 via TDM

  auto issue_tiles = [&](int k0, int buf) {
    tdm_load_2d_bf16(lds0 + (unsigned)buf * GEMM_TILE_BYTES,
                     A + (size_t)brow * K + k0,
                     GBK, GBM, (unsigned long long)K, GBK, GBM,
                     /*interval 16 DW*/ 3, /*pad 4 DW*/ 3);
    tdm_load_2d_bf16(lds0 + (unsigned)(2 + buf) * GEMM_TILE_BYTES,
                     BT + (size_t)bcol * K + k0,
                     GBK, GBN, (unsigned long long)K, GBK, bn_valid,
                     3, 3);
  };

  v8f acc[4][2];
#pragma unroll
  for (int mt = 0; mt < 4; ++mt)
#pragma unroll
    for (int nt = 0; nt < 2; ++nt)
      acc[mt][nt] = (v8f){0.f, 0.f, 0.f, 0.f, 0.f, 0.f, 0.f, 0.f};

  if (w == 0) issue_tiles(0, 0);

  int it = 0;
  for (int k0 = 0; k0 < K; k0 += GBK, ++it) {
    const int cur = it & 1;
    if (w == 0) {
      if (k0 + GBK < K) {
        issue_tiles(k0 + GBK, cur ^ 1);           // prefetch next tile pair
        __builtin_amdgcn_s_wait_tensorcnt(2);     // current pair resident
      } else {
        __builtin_amdgcn_s_wait_tensorcnt(0);
      }
    }
    __syncthreads();

    const bf16_t* As = gsm + cur * GEMM_TILE_ELEMS;
    const bf16_t* Bs = gsm + (2 + cur) * GEMM_TILE_ELEMS;
    v16bf af[4], bfg[2];
#pragma unroll
    for (int mt = 0; mt < 4; ++mt)
      af[mt] = load_a_frag(As + (wr * 64 + mt * 16) * GSK, GSK, lane);
#pragma unroll
    for (int nt = 0; nt < 2; ++nt)
      bfg[nt] = load_b_frag(Bs + (wc * 32 + nt * 16) * GSK, GSK, lane);
#pragma unroll
    for (int mt = 0; mt < 4; ++mt)
#pragma unroll
      for (int nt = 0; nt < 2; ++nt)
        acc[mt][nt] = wmma_bf16(af[mt], bfg[nt], acc[mt][nt]);
    __syncthreads();   // readers done before tile (it+2) TDM overwrites buf
  }

  const int col0 = lane & 15, half = lane >> 4;
#pragma unroll
  for (int mt = 0; mt < 4; ++mt)
#pragma unroll
    for (int nt = 0; nt < 2; ++nt) {
      int row = brow + wr * 64 + mt * 16 + half * 8;
      int col = bcol + wc * 32 + nt * 16 + col0;
      if (col < N) {
        float bv = bias[col];
#pragma unroll
        for (int r = 0; r < 8; ++r)
          C[(size_t)(row + r) * N + col] = acc[mt][nt][r] + bv;
      }
    }
}

// ---------------------------------------------------------------------------
// RoPE + split qkv(f32, SEQ x 5120) into q/k/v bf16 buffers.
// ---------------------------------------------------------------------------
__global__ void rope_split_kernel(const float* __restrict__ qkv,
                                  const float* __restrict__ cosp,
                                  const float* __restrict__ sinp,
                                  bf16_t* __restrict__ qo,
                                  bf16_t* __restrict__ ko,
                                  bf16_t* __restrict__ vo) {
  size_t idx = (size_t)blockIdx.x * blockDim.x + threadIdx.x;
  const size_t total = (size_t)SEQ * 80 * 32;
  if (idx >= total) return;
  int d  = (int)(idx & 31);
  int ht = (int)((idx >> 5) % 80);
  int s  = (int)(idx / (80 * 32));

  int base;
  if (ht < 64)      base = ht * HDIM;
  else if (ht < 72) base = QDIM + (ht - 64) * HDIM;
  else              base = QDIM + 512 + (ht - 72) * HDIM;

  const float* row = qkv + (size_t)s * NQKV;
  float x1 = row[base + d];
  float x2 = row[base + d + 32];
  float o1, o2;
  if (ht < 72) {
    float c1 = cosp[s * HDIM + d],      s1 = sinp[s * HDIM + d];
    float c2 = cosp[s * HDIM + d + 32], s2 = sinp[s * HDIM + d + 32];
    o1 = x1 * c1 - x2 * s1;
    o2 = x2 * c2 + x1 * s2;
  } else {
    o1 = x1; o2 = x2;
  }
  if (ht < 64) {
    bf16_t* p = qo + ((size_t)s * NQH + ht) * HDIM;
    p[d] = (bf16_t)o1; p[d + 32] = (bf16_t)o2;
  } else if (ht < 72) {
    bf16_t* p = ko + ((size_t)s * NKVH + (ht - 64)) * HDIM;
    p[d] = (bf16_t)o1; p[d + 32] = (bf16_t)o2;
  } else {
    bf16_t* p = vo + ((size_t)s * NKVH + (ht - 72)) * HDIM;
    p[d] = (bf16_t)o1; p[d + 32] = (bf16_t)o2;
  }
}

// ---------------------------------------------------------------------------
// Sink-window attention. Grid: (32 query blocks, 8 kv heads); 8 waves, one
// query head each. K tile arrives via TDM (strided 2D descriptor: row stride
// NKVH*HDIM elems, LDS pad to KT_STRIDE); V is staged transposed manually.
// ---------------------------------------------------------------------------
__global__ __launch_bounds__(256) void attn_kernel(
    const bf16_t* __restrict__ Q, const bf16_t* __restrict__ Kb,
    const bf16_t* __restrict__ Vb, const float* __restrict__ sinks,
    bf16_t* __restrict__ O) {
  extern __shared__ char smem_raw[];
  bf16_t* Kt = (bf16_t*)smem_raw;          // [256 keys][KT_STRIDE]
  bf16_t* Vt = Kt + 256 * KT_STRIDE;       // [64 dims][VT_STRIDE] (V transposed)
  bf16_t* Ps = Vt + 64 * VT_STRIDE;        // 8 waves x [16 rows][VT_STRIDE]

  const int blk  = blockIdx.x;
  const int kv   = blockIdx.y;
  const int tid  = threadIdx.x;
  const int lane = tid & 31;
  const int w    = tid >> 5;
  const int h    = kv * GQA + w;
  const float NEG_INF = -__builtin_huge_valf();
  const unsigned int lds0 = __builtin_amdgcn_groupstaticsize();

  // ---- K tile via Tensor Data Mover (wave 0 issues; TENSORcnt tracks) ----
  if (w == 0) {
    if (blk > 0) {
      const bf16_t* g = Kb + (((size_t)(blk * WWIN - WWIN)) * NKVH + kv) * HDIM;
      tdm_load_2d_bf16(lds0, g, HDIM, 256, (unsigned long long)(NKVH * HDIM),
                       HDIM, 256, /*interval 32 DW*/ 4, /*pad 8 DW*/ 7);
    } else {
      const bf16_t* g = Kb + (size_t)kv * HDIM;   // rows 128.. come from s=0..
      tdm_load_2d_bf16(lds0 + 128u * KT_STRIDE * 2, g, HDIM, 128,
                       (unsigned long long)(NKVH * HDIM), HDIM, 128, 4, 7);
    }
  }
  if (blk == 0) {  // zero-fill the 128 out-of-range key rows
    uint4 z = make_uint4(0u, 0u, 0u, 0u);
    for (int i = tid; i < 128 * KT_STRIDE / 8; i += 256)
      *(uint4*)(Kt + i * 8) = z;
  }

  // ---- V^T staging: thread t owns key t ----
  {
    const int key = tid;
    const long skey = (long)blk * WWIN - WWIN + key;
    if (skey >= 0) {
      const uint4* vp = (const uint4*)(Vb + ((size_t)skey * NKVH + kv) * HDIM);
      uint4 vreg[8];
#pragma unroll
      for (int i = 0; i < 8; ++i) vreg[i] = vp[i];
      const bf16_t* ve = (const bf16_t*)vreg;
#pragma unroll
      for (int d = 0; d < HDIM; ++d)
        Vt[(size_t)d * VT_STRIDE + key] = ve[d];
    } else {
      bf16_t zb = (bf16_t)0.f;
#pragma unroll
      for (int d = 0; d < HDIM; ++d)
        Vt[(size_t)d * VT_STRIDE + key] = zb;
    }
  }
  if (w == 0) __builtin_amdgcn_s_wait_tensorcnt(0);
  __syncthreads();

  const float skv = sinks[h];
  bf16_t* Pw = Ps + (size_t)w * 16 * VT_STRIDE;
  const int s0 = blk * WWIN;
  const int col0 = lane & 15, half = lane >> 4;

  for (int mc = 0; mc < 8; ++mc) {
    const int qrow0 = s0 + mc * 16;
    const bf16_t* qb = Q + (size_t)qrow0 * QDIM + h * HDIM;
    if (mc + 1 < 8) __builtin_prefetch(qb + 16 * QDIM, 0, 1);
    v16bf qa0 = load_a_frag(qb, QDIM, lane);
    v16bf qa1 = load_a_frag(qb + 32, QDIM, lane);

    // ---- scores ----
    v8f sc[16];
#pragma unroll
    for (int nt = 0; nt < 16; ++nt) {
      v8f c = (v8f){0.f, 0.f, 0.f, 0.f, 0.f, 0.f, 0.f, 0.f};
      v16bf b0 = load_b_frag(Kt + nt * 16 * KT_STRIDE, KT_STRIDE, lane);
      v16bf b1 = load_b_frag(Kt + nt * 16 * KT_STRIDE + 32, KT_STRIDE, lane);
      c = wmma_bf16(qa0, b0, c);
      c = wmma_bf16(qa1, b1, c);
      sc[nt] = c;
    }

    // ---- mask + scale ----
#pragma unroll
    for (int nt = 0; nt < 16; ++nt) {
      const int kj = nt * 16 + col0;
#pragma unroll
      for (int r = 0; r < 8; ++r) {
        const int qi = mc * 16 + half * 8 + r;
        const bool valid = (kj > qi) && (kj <= qi + WWIN) && (blk > 0 || kj >= WWIN);
        sc[nt][r] = valid ? sc[nt][r] * ATT_SCALE : NEG_INF;
      }
    }

    // ---- softmax with sink ----
    float inv[8], rmax[8];
#pragma unroll
    for (int r = 0; r < 8; ++r) {
      float m = NEG_INF;
#pragma unroll
      for (int nt = 0; nt < 16; ++nt) m = fmaxf(m, sc[nt][r]);
      m = fmaxf(m, __shfl_xor(m, 1, 32));
      m = fmaxf(m, __shfl_xor(m, 2, 32));
      m = fmaxf(m, __shfl_xor(m, 4, 32));
      m = fmaxf(m, __shfl_xor(m, 8, 32));
      m = fmaxf(m, skv);
      rmax[r] = m;
      float s = 0.f;
#pragma unroll
      for (int nt = 0; nt < 16; ++nt) {
        float e = __expf(sc[nt][r] - m);
        sc[nt][r] = e;
        s += e;
      }
      s += __shfl_xor(s, 1, 32);
      s += __shfl_xor(s, 2, 32);
      s += __shfl_xor(s, 4, 32);
      s += __shfl_xor(s, 8, 32);
      s += __expf(skv - m);
      inv[r] = 1.f / s;
    }

    // ---- P -> LDS (C-layout -> A-layout crossover) ----
#pragma unroll
    for (int nt = 0; nt < 16; ++nt)
#pragma unroll
      for (int r = 0; r < 8; ++r)
        Pw[(size_t)(half * 8 + r) * VT_STRIDE + nt * 16 + col0] =
            (bf16_t)(sc[nt][r] * inv[r]);
    asm volatile("s_wait_dscnt 0" ::: "memory");

    // ---- PV ----
#pragma unroll
    for (int no = 0; no < 4; ++no) {
      v8f o = (v8f){0.f, 0.f, 0.f, 0.f, 0.f, 0.f, 0.f, 0.f};
#pragma unroll
      for (int kt = 0; kt < 8; ++kt) {
        v16bf pa = load_a_frag(Pw + kt * 32, VT_STRIDE, lane);
        v16bf vb = load_b_frag(Vt + (size_t)(no * 16) * VT_STRIDE + kt * 32,
                               VT_STRIDE, lane);
        o = wmma_bf16(pa, vb, o);
      }
#pragma unroll
      for (int r = 0; r < 8; ++r) {
        size_t row = (size_t)(qrow0 + half * 8 + r);
        O[row * QDIM + h * HDIM + no * 16 + col0] = (bf16_t)o[r];
      }
    }
  }
}

// ---------------------------------------------------------------------------
// Host-side orchestration
// ---------------------------------------------------------------------------
extern "C" void kernel_launch(void* const* d_in, const int* in_sizes, int n_in,
                              void* d_out, int out_size, void* d_ws, size_t ws_size,
                              hipStream_t stream) {
  (void)in_sizes; (void)n_in; (void)out_size; (void)ws_size;
  const float* hs    = (const float*)d_in[0];
  const float* cosp  = (const float*)d_in[1];
  const float* sinp  = (const float*)d_in[2];
  const float* wqkv  = (const float*)d_in[3];
  const float* bqkv  = (const float*)d_in[4];
  const float* wo    = (const float*)d_in[5];
  const float* bo    = (const float*)d_in[6];
  const float* sinks = (const float*)d_in[7];
  float* out = (float*)d_out;

  char* ws = (char*)d_ws;
  size_t off = 0;
  bf16_t* hs_bf   = (bf16_t*)(ws + off); off += (size_t)SEQ * HID * 2;
  bf16_t* wqkvT   = (bf16_t*)(ws + off); off += (size_t)NQKV * HID * 2;
  bf16_t* woT     = (bf16_t*)(ws + off); off += (size_t)HID * QDIM * 2;
  float*  qkv     = (float*) (ws + off); off += (size_t)SEQ * NQKV * 4;
  bf16_t* q_bf    = (bf16_t*)(ws + off); off += (size_t)SEQ * QDIM * 2;
  bf16_t* k_bf    = (bf16_t*)(ws + off); off += (size_t)SEQ * NKVH * HDIM * 2;
  bf16_t* v_bf    = (bf16_t*)(ws + off); off += (size_t)SEQ * NKVH * HDIM * 2;
  bf16_t* attn_bf = (bf16_t*)(ws + off); off += (size_t)SEQ * QDIM * 2;

  // 1) conversions
  convert_f32_bf16_kernel<<<4096, 256, 0, stream>>>(hs, hs_bf, (size_t)SEQ * HID);
  transpose_bf16_kernel<<<dim3(NQKV / 32, HID / 32), 256, 0, stream>>>(
      wqkv, wqkvT, HID, NQKV);
  transpose_bf16_kernel<<<dim3((HID + 31) / 32, QDIM / 32), 256, 0, stream>>>(
      wo, woT, QDIM, HID);

  // 2) QKV GEMM (TDM double-buffered)
  (void)hipFuncSetAttribute((const void*)gemm_bf16_kernel,
                            hipFuncAttributeMaxDynamicSharedMemorySize, GEMM_SMEM);
  gemm_bf16_kernel<<<dim3(NQKV / GBN, SEQ / GBM), 256, GEMM_SMEM, stream>>>(
      hs_bf, wqkvT, bqkv, qkv, SEQ, NQKV, HID);

  // 3) RoPE + split
  {
    size_t total = (size_t)SEQ * 80 * 32;
    rope_split_kernel<<<(int)((total + 255) / 256), 256, 0, stream>>>(
        qkv, cosp, sinp, q_bf, k_bf, v_bf);
  }

  // 4) attention
  {
    const int smem = (256 * KT_STRIDE + 64 * VT_STRIDE + 8 * 16 * VT_STRIDE) * 2;
    (void)hipFuncSetAttribute((const void*)attn_kernel,
                              hipFuncAttributeMaxDynamicSharedMemorySize, smem);
    attn_kernel<<<dim3(SEQ / WWIN, NKVH), 256, smem, stream>>>(
        q_bf, k_bf, v_bf, sinks, attn_bf);
  }

  // 5) output GEMM (TDM zero-fills the N=2880 edge rows)
  gemm_bf16_kernel<<<dim3((HID + GBN - 1) / GBN, SEQ / GBM), 256, GEMM_SMEM, stream>>>(
      attn_bf, woT, bo, out, SEQ, HID, QDIM);
}
